// TriadKuramotoLayer_15925738734071
// MI455X (gfx1250) — compile-verified, matrix operand order checked by reading, not run on hardware
//
#include <hip/hip_runtime.h>
#include <math.h>

typedef __attribute__((ext_vector_type(16))) _Float16 v16h;
typedef __attribute__((ext_vector_type(8)))  _Float16 v8h;
typedef __attribute__((ext_vector_type(8)))  float    v8f;

#define NOSC   2048
#define NBATCH 16
#define KDT    0.1f
#define NSTEPS 10
#define WPB    4   // waves per block in the GEMM kernel

// ---------------------------------------------------------------------------
// Kernel 1: K_eff = K * (1 + triad_climb_mod), converted to f16 (once).
// K_eff (8 MB in f16) stays resident in the 192 MB L2 for all 10 steps.
// ---------------------------------------------------------------------------
__global__ __launch_bounds__(256)
void keff_f16_kernel(const float* __restrict__ K,
                     const float* __restrict__ mod,
                     _Float16* __restrict__ Kh) {
  const int idx = blockIdx.x * blockDim.x + threadIdx.x;
  const float scale = 1.0f + mod[0];
  Kh[idx] = (_Float16)(K[idx] * scale);
}

// ---------------------------------------------------------------------------
// Kernel 2 (per step): build f16 sin(theta), cos(theta) panels, layout [B][N]
// (batch-major, oscillator-contiguous) so WMMA B-operand loads are contiguous.
// ---------------------------------------------------------------------------
__global__ __launch_bounds__(256)
void trig_kernel(const float* __restrict__ theta,
                 _Float16* __restrict__ SB,
                 _Float16* __restrict__ CB) {
  const int idx = blockIdx.x * blockDim.x + threadIdx.x;
  const float t = theta[idx];
  SB[idx] = (_Float16)sinf(t);
  CB[idx] = (_Float16)cosf(t);
}

// ---------------------------------------------------------------------------
// Kernel 3 (per step): fused dual-GEMM + theta update.
// Each wave computes one 16(row)x16(batch) output tile:
//   accS = K_eff[tile,:] @ sin(theta)[:,b]   (f32 accum, f16 inputs)
//   accC = K_eff[tile,:] @ cos(theta)[:,b]
// then applies the Kuramoto update + wrap for its 256 (i,b) elements using
// the WMMA C/D VGPR layout (lane L, vgpr v -> row tile+v+8*(L>=16), col L%16).
//
// 16-bit A-operand layout (ISA 7.12.2): lane L holds row M=L%16; halfs
// {kb..kb+7} in v[0..3] and {kb+16..kb+23} in v[4..7], kb = (L<16?0:8).
// B-operand mirrors this with column n = L%16. Both load as two b128s.
// ---------------------------------------------------------------------------
__global__ __launch_bounds__(32 * WPB)
void gemm_update_kernel(const _Float16* __restrict__ Kh,   // [N][N] f16
                        const _Float16* __restrict__ SB,   // [B][N] f16
                        const _Float16* __restrict__ CB,   // [B][N] f16
                        const float* __restrict__ theta_src, // [B][N]
                        const float* __restrict__ omega,     // [N]
                        const float* __restrict__ Kg,        // scalar
                        float* __restrict__ theta_dst) {     // [B][N]
  const int lane    = threadIdx.x & 31;
  const int wave    = threadIdx.x >> 5;
  const int rowTile = (blockIdx.x * WPB + wave) << 4;   // 16 output rows
  const int rsub    = lane & 15;                        // row (A) / col (B)
  const int hi      = lane >> 4;                        // half-wave select
  const int kbase   = hi << 3;                          // K sub-offset {0,8}

  const _Float16* __restrict__ Arow = Kh + (size_t)(rowTile + rsub) * NOSC + kbase;
  const _Float16* __restrict__ Bs   = SB + (size_t)rsub * NOSC + kbase;
  const _Float16* __restrict__ Bc   = CB + (size_t)rsub * NOSC + kbase;

  v8f accS = {};
  v8f accC = {};

  for (int kt = 0; kt < NOSC; kt += 32) {
    const v8h alo = *(const v8h*)(Arow + kt);
    const v8h ahi = *(const v8h*)(Arow + kt + 16);
    const v8h slo = *(const v8h*)(Bs + kt);
    const v8h shi = *(const v8h*)(Bs + kt + 16);
    const v8h clo = *(const v8h*)(Bc + kt);
    const v8h chi = *(const v8h*)(Bc + kt + 16);
    v16h a, bs, bc;
#pragma unroll
    for (int i = 0; i < 8; ++i) {
      a[i]  = alo[i];  a[i + 8]  = ahi[i];
      bs[i] = slo[i];  bs[i + 8] = shi[i];
      bc[i] = clo[i];  bc[i + 8] = chi[i];
    }
    // 8 args: (neg_a, A, neg_b, B, c_mod, C, reuse_a, reuse_b)
    accS = __builtin_amdgcn_wmma_f32_16x16x32_f16(false, a, false, bs,
                                                  (short)0, accS, false, false);
    accC = __builtin_amdgcn_wmma_f32_16x16x32_f16(false, a, false, bc,
                                                  (short)0, accC, false, false);
  }

  // Kuramoto update + wrap for this lane's 8 C/D elements.
  const float kg = Kg[0] * (1.0f / (float)NOSC);
  const int b = rsub;  // batch = output column
#pragma unroll
  for (int v = 0; v < 8; ++v) {
    const int i = rowTile + (hi << 3) + v;       // oscillator row
    const size_t off = (size_t)b * NOSC + i;
    float th = theta_src[off];
    const float s = sinf(th);
    const float c = cosf(th);
    const float dth = omega[i] + kg * (c * accS[v] - s * accC[v]);
    th += KDT * dth;
    theta_dst[off] = atan2f(sinf(th), cosf(th)); // wrap to (-pi, pi]
  }
}

// ---------------------------------------------------------------------------
// Kernel 4: coherence[b] = sqrt(mean(cos th)^2 + mean(sin th)^2)
// ---------------------------------------------------------------------------
__global__ __launch_bounds__(256)
void coherence_kernel(const float* __restrict__ theta,
                      float* __restrict__ coh) {
  __shared__ float sc[256];
  __shared__ float ss[256];
  const int b = blockIdx.x;
  const int t = threadIdx.x;
  float cs = 0.0f, sn = 0.0f;
  for (int i = t; i < NOSC; i += 256) {
    const float th = theta[(size_t)b * NOSC + i];
    cs += cosf(th);
    sn += sinf(th);
  }
  sc[t] = cs; ss[t] = sn;
  __syncthreads();
  for (int ofs = 128; ofs > 0; ofs >>= 1) {
    if (t < ofs) { sc[t] += sc[t + ofs]; ss[t] += ss[t + ofs]; }
    __syncthreads();
  }
  if (t == 0) {
    const float mc = sc[0] * (1.0f / (float)NOSC);
    const float ms = ss[0] * (1.0f / (float)NOSC);
    coh[b] = sqrtf(mc * mc + ms * ms);
  }
}

// ---------------------------------------------------------------------------
// Host launcher. Workspace: Kh (N*N f16 = 8 MB) | SB (B*N f16) | CB (B*N f16).
// Step 0 reads theta from d_in, writes d_out; steps 1..9 read/write d_out.
// All launches on `stream`; no allocation / sync (graph-capture safe).
// ---------------------------------------------------------------------------
extern "C" void kernel_launch(void* const* d_in, const int* in_sizes, int n_in,
                              void* d_out, int out_size, void* d_ws, size_t ws_size,
                              hipStream_t stream) {
  (void)in_sizes; (void)n_in; (void)out_size; (void)ws_size;

  const float* theta0 = (const float*)d_in[0];  // [B,N]
  const float* K      = (const float*)d_in[1];  // [N,N]
  const float* omega  = (const float*)d_in[2];  // [N]
  const float* Kg     = (const float*)d_in[3];  // scalar
  const float* mod    = (const float*)d_in[4];  // scalar

  float* out_theta = (float*)d_out;                      // [B,N]
  float* out_coh   = out_theta + (size_t)NBATCH * NOSC;  // [B]

  _Float16* Kh = (_Float16*)d_ws;
  _Float16* SB = Kh + (size_t)NOSC * NOSC;
  _Float16* CB = SB + (size_t)NBATCH * NOSC;

  keff_f16_kernel<<<(NOSC * NOSC) / 256, 256, 0, stream>>>(K, mod, Kh);

  const float* src = theta0;
  for (int s = 0; s < NSTEPS; ++s) {
    trig_kernel<<<(NBATCH * NOSC) / 256, 256, 0, stream>>>(src, SB, CB);
    gemm_update_kernel<<<NOSC / (16 * WPB), 32 * WPB, 0, stream>>>(
        Kh, SB, CB, src, omega, Kg, out_theta);
    src = out_theta;
  }

  coherence_kernel<<<NBATCH, 256, 0, stream>>>(out_theta, out_coh);
}
